// Kpe_81587198755186
// MI455X (gfx1250) — matrix operand order repaired and checked by reading, compiled, not run on hardware
//
#include <hip/hip_runtime.h>
#include <hip/hip_bf16.h>

// Problem constants (from reference)
#define BB 64
#define SS 512
#define HH 768
#define CC 512
#define MM 512
#define NL 5
#define ROWS (BB * SS)   // 32768

// Use inline asm for the CDNA5 async global->LDS path (clang builtin has a
// descriptor-style signature on this toolchain; asm maps 1:1 to the ISA).
#define USE_ASYNC_ASM 1

typedef __attribute__((ext_vector_type(16))) __bf16 v16bf;
typedef __attribute__((ext_vector_type(8)))  float  v8f;

union Frag16 { v16bf v; uint4 u[2]; };

__device__ __forceinline__ unsigned short f2bf(float f) {
  unsigned u = __float_as_uint(f);
  unsigned r = u + 0x7FFFu + ((u >> 16) & 1u);   // round-to-nearest-even
  return (unsigned short)(r >> 16);
}
__device__ __forceinline__ unsigned pk2(float a, float b) {
  return (unsigned)f2bf(a) | ((unsigned)f2bf(b) << 16);
}

// generic (__shared__) pointer -> 32-bit LDS byte address
__device__ __forceinline__ unsigned toLds(const void* p) {
  return (unsigned)(size_t)(__attribute__((address_space(3))) const void*)p;
}

// async DMA: 16 bytes per lane, global -> LDS (tracked by ASYNCcnt)
__device__ __forceinline__ void asyncB128(const unsigned short* g, unsigned short* l) {
#if USE_ASYNC_ASM
  asm volatile("global_load_async_to_lds_b128 %0, %1, off"
               :: "v"(toLds(l)), "v"((unsigned long long)(size_t)g)
               : "memory");
#else
  *(uint4*)l = *(const uint4*)g;
#endif
}
__device__ __forceinline__ void waitAsync() {
#if USE_ASYNC_ASM
  asm volatile("s_wait_asynccnt 0" ::: "memory");
#endif
}

// ---------------------------------------------------------------- init
__global__ void initK(float* accums) {
  accums[0] = 0.f;  // ce_sum
  accums[1] = 0.f;  // ce_cnt
  accums[2] = 0.f;  // rank_sum
}

// ---------------------------------------------------------------- conv_w f32 -> bf16 (one shot)
__global__ void convK(const float* __restrict__ w, unsigned short* __restrict__ wbf) {
  int i = (blockIdx.x * 256 + threadIdx.x) * 8;
  float4 f0 = *(const float4*)(w + i);
  float4 f1 = *(const float4*)(w + i + 4);
  uint4 o;
  o.x = pk2(f0.x, f0.y); o.y = pk2(f0.z, f0.w);
  o.z = pk2(f1.x, f1.y); o.w = pk2(f1.z, f1.w);
  *(uint4*)(wbf + i) = o;
}

// ---------------------------------------------------------------- ragged compaction scan
__global__ void scanK(const int* __restrict__ valid_ids, int* __restrict__ g) {
  __shared__ int pref[SS];
  int b = blockIdx.x, t = threadIdx.x;
  int v = (valid_ids[b * SS + t] == 1) ? 1 : 0;
  pref[t] = v;
  __syncthreads();
  for (int off = 1; off < SS; off <<= 1) {
    int x = (t >= off) ? pref[t - off] : 0;
    __syncthreads();
    pref[t] += x;
    __syncthreads();
  }
  int count = pref[SS - 1];
  int p = pref[t] - v;            // exclusive prefix
  if (v) g[b * SS + p] = t;       // p-th valid token came from position t
  if (t >= count) g[b * SS + t] = -1;
}

// ---------------------------------------------------------------- fused GEMM + epilogue
// block = 256 threads (8 waves), owns 32 rows x 512 cols of cnn.
// Wave w owns both 16-row tiles x 4 col-tiles (cols w*64..w*64+63):
//   per K-step: 2 A-frags + 4 B-frags (12 ds_b128) -> 8 WMMAs.
// Double-buffered LDS; B staged via async global->LDS DMA; 1 barrier / K-step.
__global__ void __launch_bounds__(256)
gemmK(const float* __restrict__ enc, const float* __restrict__ vout,
      const int* __restrict__ g, const unsigned short* __restrict__ wbf,
      const float* __restrict__ conv_b,
      const float* __restrict__ cls_w,  const float* __restrict__ cls_b,
      const float* __restrict__ chunk_w, const float* __restrict__ chunk_b,
      const int* __restrict__ chunk_mask, const int* __restrict__ chunk_labels,
      float* __restrict__ d_scores, float* __restrict__ accums) {
  __shared__ unsigned short As[2][32][32];     // 4 KB   (rows x K)
  __shared__ unsigned short Bs[2][CC][32];     // 64 KB  (cols x K, K contiguous)
  __shared__ const float*   rowp[32];
  __shared__ float          red[32 * 6];       // per-row: score + 5 chunk partials

  const int tid = threadIdx.x;
  const int rowBase = blockIdx.x * 32;

  if (tid < 32) {
    int row = rowBase + tid;
    int gi = g[row];
    int b = row >> 9;
    rowp[tid] = (gi >= 0) ? (enc + (size_t)(b * SS + gi) * HH)
                          : (vout + (size_t)row * HH);
  }
  if (tid < 192) red[tid] = 0.f;
  __syncthreads();

  // staging coordinates
  const int ra  = tid >> 3;             // A: row 0..31
  const int k0a = (tid & 7) * 4;        // A: 4 K-values
  const float* rowA = rowp[ra];
  const int c0 = tid, c1 = tid + 256;   // B: 2 columns per thread
  const unsigned short* wc0 = wbf + (size_t)c0 * HH;
  const unsigned short* wc1 = wbf + (size_t)c1 * HH;

  // compute coordinates
  const int wv = tid >> 5, lane = tid & 31;
  const int half = lane >> 4, rl = lane & 15;

  v8f acc[2][4];
#pragma unroll
  for (int i2 = 0; i2 < 2; ++i2)
#pragma unroll
    for (int jt = 0; jt < 4; ++jt) acc[i2][jt] = (v8f)0.0f;

  // ---- prologue: stage K-tile 0 into buffer 0
  {
    const float4 f = *(const float4*)(rowA + k0a);
    uint2 u; u.x = pk2(f.x, f.y); u.y = pk2(f.z, f.w);
    *(uint2*)&As[0][ra][k0a] = u;
#pragma unroll
    for (int q = 0; q < 4; ++q) {
      asyncB128(wc0 + q * 8, &Bs[0][c0][q * 8]);
      asyncB128(wc1 + q * 8, &Bs[0][c1][q * 8]);
    }
    waitAsync();
  }
  __syncthreads();

  int buf = 0;
  for (int kb = 0; kb < HH; kb += 32) {
    const int nkb = kb + 32;
    const int nb = buf ^ 1;
    float4 fa;
    if (nkb < HH) {
      // issue next-tile staging (async DMA for B, register load for A)
#pragma unroll
      for (int q = 0; q < 4; ++q) {
        asyncB128(wc0 + nkb + q * 8, &Bs[nb][c0][q * 8]);
        asyncB128(wc1 + nkb + q * 8, &Bs[nb][c1][q * 8]);
      }
      fa = *(const float4*)(rowA + nkb + k0a);
    }

    // ---- compute current buffer: 2 A-frags, 4 B-frags, 8 WMMAs
    Frag16 a0, a1;
    a0.u[0] = *(const uint4*)&As[buf][rl][half * 8];
    a0.u[1] = *(const uint4*)&As[buf][rl][16 + half * 8];
    a1.u[0] = *(const uint4*)&As[buf][16 + rl][half * 8];
    a1.u[1] = *(const uint4*)&As[buf][16 + rl][16 + half * 8];
#pragma unroll
    for (int jt = 0; jt < 4; ++jt) {
      int col = (wv * 4 + jt) * 16 + rl;
      Frag16 bfr;
      bfr.u[0] = *(const uint4*)&Bs[buf][col][half * 8];
      bfr.u[1] = *(const uint4*)&Bs[buf][col][16 + half * 8];
      acc[0][jt] = __builtin_amdgcn_wmma_f32_16x16x32_bf16(
          false, a0.v, false, bfr.v, (short)0, acc[0][jt], false, false);
      acc[1][jt] = __builtin_amdgcn_wmma_f32_16x16x32_bf16(
          false, a1.v, false, bfr.v, (short)0, acc[1][jt], false, false);
    }

    // ---- store next-tile A registers into the other buffer
    if (nkb < HH) {
      uint2 u; u.x = pk2(fa.x, fa.y); u.y = pk2(fa.z, fa.w);
      *(uint2*)&As[nb][ra][k0a] = u;
    }
    waitAsync();
    __syncthreads();
    buf = nb;
  }

  // ---- epilogue: bias + relu, fused row reductions (cls score, 5 chunk logits)
  // process the two 16-row tiles sequentially to keep register pressure low
#pragma unroll
  for (int i2 = 0; i2 < 2; ++i2) {
    float sc[8];
    float c5[8][NL];
#pragma unroll
    for (int v = 0; v < 8; ++v) {
      sc[v] = 0.f;
#pragma unroll
      for (int l = 0; l < NL; ++l) c5[v][l] = 0.f;
    }
#pragma unroll
    for (int jt = 0; jt < 4; ++jt) {
      int col = (wv * 4 + jt) * 16 + rl;
      float cwb = conv_b[col];
      float clw = cls_w[col];
      float chw[NL];
#pragma unroll
      for (int l = 0; l < NL; ++l) chw[l] = chunk_w[l * CC + col];
#pragma unroll
      for (int v = 0; v < 8; ++v) {
        float val = fmaxf(acc[i2][jt][v] + cwb, 0.f);   // relu(cnn)
        sc[v] += val * clw;
#pragma unroll
        for (int l = 0; l < NL; ++l) c5[v][l] += val * chw[l];
      }
    }
#pragma unroll
    for (int v = 0; v < 8; ++v) {
#pragma unroll
      for (int off = 1; off < 16; off <<= 1) {
        sc[v] += __shfl_xor(sc[v], off, 32);
#pragma unroll
        for (int l = 0; l < NL; ++l) c5[v][l] += __shfl_xor(c5[v][l], off, 32);
      }
      if (rl == 0) {
        int row_local = i2 * 16 + half * 8 + v;
        atomicAdd(&red[row_local * 6], sc[v]);
#pragma unroll
        for (int l = 0; l < NL; ++l) atomicAdd(&red[row_local * 6 + 1 + l], c5[v][l]);
      }
    }
  }
  __syncthreads();

  if (tid < 32) {
    int row = rowBase + tid;
    float score = red[tid * 6] + cls_b[0];
    d_scores[row] = score;
    if (chunk_mask[row] != -1) {
      float lg[NL], mx = -__builtin_inff();
#pragma unroll
      for (int l = 0; l < NL; ++l) { lg[l] = red[tid * 6 + 1 + l] + chunk_b[l]; mx = fmaxf(mx, lg[l]); }
      float se = 0.f;
#pragma unroll
      for (int l = 0; l < NL; ++l) se += __expf(lg[l] - mx);
      float lse = mx + __logf(se);
      int lab = chunk_labels[row];
      lab = (lab < 0) ? 0 : (lab > NL - 1 ? NL - 1 : lab);
      atomicAdd(&accums[0], lse - lg[lab]);   // ce sum
      atomicAdd(&accums[1], 1.0f);            // ce count
    }
  }
}

// ---------------------------------------------------------------- masked max -> total_scores
__global__ void maxK(const float* __restrict__ scores, const int* __restrict__ amask,
                     float* __restrict__ out /* d_out + 1 */) {
  __shared__ float ssc[SS];
  __shared__ float wred[8];
  int b = blockIdx.x >> 3;
  int mBase = (blockIdx.x & 7) * 64;
  int tid = threadIdx.x;
  ssc[tid] = scores[b * SS + tid];
  ssc[tid + 256] = scores[b * SS + tid + 256];
  __syncthreads();
  for (int mm = 0; mm < 64; ++mm) {
    int m = mBase + mm;
    const int* mp = amask + (size_t)(b * MM + m) * SS;
    float v = mp[tid] ? -__builtin_inff() : ssc[tid];
    float v2 = mp[tid + 256] ? -__builtin_inff() : ssc[tid + 256];
    v = fmaxf(v, v2);
#pragma unroll
    for (int off = 1; off < 32; off <<= 1) v = fmaxf(v, __shfl_xor(v, off, 32));
    if ((tid & 31) == 0) wred[tid >> 5] = v;
    __syncthreads();
    if (tid == 0) {
      float r = wred[0];
#pragma unroll
      for (int w = 1; w < 8; ++w) r = fmaxf(r, wred[w]);
      out[b * MM + m] = r;
    }
    __syncthreads();
  }
}

// ---------------------------------------------------------------- margin ranking loss
__global__ void rankK(const float* __restrict__ tscores, const int* __restrict__ labels,
                      float* __restrict__ accums) {
  __shared__ float ts[MM];
  __shared__ int   lb[MM];
  __shared__ float rsum[8];
  __shared__ int   rnp[8], rnn[8];
  int b = blockIdx.x, tid = threadIdx.x;
  ts[tid] = tscores[b * MM + tid];      ts[tid + 256] = tscores[b * MM + tid + 256];
  lb[tid] = labels[b * MM + tid];       lb[tid + 256] = labels[b * MM + tid + 256];
  __syncthreads();
  float sum = 0.f; int np = 0, nn = 0;
  for (int i = tid; i < MM; i += 256) {
    int li = lb[i];
    np += (li == 1);
    nn += (li == -1);
    if (li == 1) {
      float ti = ts[i];
      for (int j = 0; j < MM; ++j)
        if (lb[j] == -1) sum += fmaxf(0.f, ts[j] - ti + 1.0f);
    }
  }
#pragma unroll
  for (int off = 1; off < 32; off <<= 1) {
    sum += __shfl_xor(sum, off, 32);
    np  += __shfl_xor(np,  off, 32);
    nn  += __shfl_xor(nn,  off, 32);
  }
  if ((tid & 31) == 0) { rsum[tid >> 5] = sum; rnp[tid >> 5] = np; rnn[tid >> 5] = nn; }
  __syncthreads();
  if (tid == 0) {
    float s = 0.f; int tp = 0, tn = 0;
#pragma unroll
    for (int w = 0; w < 8; ++w) { s += rsum[w]; tp += rnp[w]; tn += rnn[w]; }
    int npairs = tp * tn;
    float loss_b = (tn == 0) ? 0.f : s / (float)(npairs > 1 ? npairs : 1);
    atomicAdd(&accums[2], loss_b * (1.0f / (float)BB));
  }
}

// ---------------------------------------------------------------- finalize
__global__ void finK(const float* __restrict__ accums, float* __restrict__ out) {
  out[0] = accums[2] + accums[0] / fmaxf(accums[1], 1.0f);
}

extern "C" void kernel_launch(void* const* d_in, const int* in_sizes, int n_in,
                              void* d_out, int out_size, void* d_ws, size_t ws_size,
                              hipStream_t stream) {
  const float* enc          = (const float*)d_in[0];   // [B,S,H]
  const int*   valid_ids    = (const int*)  d_in[1];   // [B,S]
  const float* vout         = (const float*)d_in[2];   // [B,S,H]
  const int*   amask        = (const int*)  d_in[3];   // [B,M,S] bool
  const int*   labels       = (const int*)  d_in[4];   // [B,M]
  const int*   chunk_labels = (const int*)  d_in[5];   // [B,S]
  const int*   chunk_mask   = (const int*)  d_in[6];   // [B,S]
  const float* conv_w       = (const float*)d_in[7];   // [C,H]
  const float* conv_b       = (const float*)d_in[8];   // [C]
  const float* cls_w        = (const float*)d_in[9];   // [C]
  const float* cls_b        = (const float*)d_in[10];  // scalar
  const float* chunk_w      = (const float*)d_in[11];  // [NL,C]
  const float* chunk_b      = (const float*)d_in[12];  // [NL]

  float* out = (float*)d_out;            // [0]=loss, [1..] = total_scores [B,M]
  char*  ws  = (char*)d_ws;
  int*            gIdx    = (int*)ws;                          // 32768 ints
  float*          scores  = (float*)(ws + 131072);             // 32768 floats
  float*          accums  = (float*)(ws + 262144);             // ce_sum, ce_cnt, rank_sum
  unsigned short* wbf     = (unsigned short*)(ws + 327680);    // conv_w bf16 [C][H]

  initK<<<1, 1, 0, stream>>>(accums);
  convK<<<(CC * HH) / (256 * 8), 256, 0, stream>>>(conv_w, wbf);
  scanK<<<BB, SS, 0, stream>>>(valid_ids, gIdx);
  gemmK<<<ROWS / 32, 256, 0, stream>>>(enc, vout, gIdx, wbf, conv_b, cls_w, cls_b,
                                       chunk_w, chunk_b, chunk_mask, chunk_labels,
                                       scores, accums);
  maxK<<<BB * 8, 256, 0, stream>>>(scores, amask, out + 1);
  rankK<<<BB, 256, 0, stream>>>(out + 1, labels, accums);
  finK<<<1, 1, 0, stream>>>(accums, out);
}
